// LanguageModel_50921132261988
// MI455X (gfx1250) — matrix operand order repaired and checked by reading, not compiled, not run on hardware
//
#include <hip/hip_runtime.h>
#include <hip/hip_bf16.h>
#include <math.h>

// ---------------- model dims ----------------
#define VOCAB  50257
#define VPAD   50272           // 3142 * 16
#define NTHEAD 3142            // VPAD / 16
#define NBLKX  393             // ceil(NTHEAD / 8) head-GEMM blocks along N
#define EMB    384
#define SEQ    2048
#define NB     4
#define MTOT   (NB * SEQ)      // 8192

typedef __attribute__((ext_vector_type(16))) __bf16 bf16x16;
typedef __attribute__((ext_vector_type(8)))  __bf16 bf16x8;
typedef __attribute__((ext_vector_type(8)))  float  f32x8;

union ABFrag { bf16x16 v; bf16x8 h[2]; };

// ---- WMMA helpers -----------------------------------------------------------
// A-matrix 16x32 bf16 (ISA 7.12.2): lane L: M = L%16, half = L/16.
// element e (0..7):  K = half*8 + e          (contiguous run of 8)
// element e (8..15): K = 16 + half*8 + (e-8) (contiguous run of 8)
__device__ inline bf16x16 load_a_bf16(const __bf16* base, int ldk, int m0, int k0, int lane) {
  int row  = m0 + (lane & 15);
  int half = lane >> 4;
  const __bf16* p = base + (size_t)row * ldk + k0 + half * 8;
  ABFrag f;
  f.h[0] = *(const bf16x8*)(p);
  f.h[1] = *(const bf16x8*)(p + 16);
  return f.v;
}

// B-matrix 32x16 bf16: lane L: N = L%16, e<8: K = (L/16)*8 + e, e>=8: K = 16 + (L/16)*8 + (e-8).
// `base` stores B transposed as [N][ldk] so the 16 K values are contiguous.
__device__ inline bf16x16 load_b_bf16(const __bf16* base, int ldk, int n0, int k0, int lane) {
  int nn   = n0 + (lane & 15);
  int half = lane >> 4;
  const __bf16* p = base + (size_t)nn * ldk + k0 + half * 8;
  ABFrag f;
  f.h[0] = *(const bf16x8*)(p);
  f.h[1] = *(const bf16x8*)(p + 16);
  return f.v;
}

__device__ inline f32x8 wmma_bf16(bf16x16 a, bf16x16 b, f32x8 c) {
  // (neg_a, A, neg_b, B, c_mod, C, reuse_a, reuse_b)
  return __builtin_amdgcn_wmma_f32_16x16x32_bf16(false, a, false, b, (short)0, c, false, false);
}

// merge (m,l) logsumexp partial pairs; neutral element is (-inf, 0)
__device__ inline void lse_merge(float& m, float& l, float m2, float l2) {
  float M = fmaxf(m, m2);
  if (M == -INFINITY) { m = M; l = 0.f; return; }
  l = l * __expf(m - M) + l2 * __expf(m2 - M);
  m = M;
}

// ---- 1. embedding: x = tok_emb[idx] + pos_emb -------------------------------
__global__ void k_embed(const int* __restrict__ idx, const float* __restrict__ tok,
                        const float* __restrict__ pos, __bf16* __restrict__ xb) {
  size_t i = (size_t)blockIdx.x * blockDim.x + threadIdx.x;
  if (i >= (size_t)MTOT * EMB) return;
  int m = (int)(i / EMB), c = (int)(i % EMB);
  int t = m & (SEQ - 1);
  int tk = idx[m];
  xb[i] = (__bf16)(tok[(size_t)tk * EMB + c] + pos[(size_t)t * EMB + c]);
}

// ---- 2. transpose-convert square weights (stored [in,out] -> bf16 [out][in])
__global__ void k_tw(const float* __restrict__ w, __bf16* __restrict__ wt) {
  int i = blockIdx.x * blockDim.x + threadIdx.x;
  if (i >= EMB * EMB) return;
  int n = i / EMB, k = i % EMB;
  wt[i] = (__bf16)w[(size_t)k * EMB + n];
}

// ---- 3. transpose-convert W_head ([384,50257] -> bf16 [50272][384], padded)
__global__ void k_twhead(const float* __restrict__ w, __bf16* __restrict__ wt) {
  size_t i = (size_t)blockIdx.x * blockDim.x + threadIdx.x;
  if (i >= (size_t)VPAD * EMB) return;
  size_t n = i / EMB; int k = (int)(i % EMB);
  wt[i] = (n < VOCAB) ? (__bf16)w[(size_t)k * VOCAB + n] : (__bf16)0.0f;
}

// ---- 4. QKV GEMM: [8192x384]x[384x384], one wave per 16x16 tile -------------
__global__ void k_qkv(const __bf16* __restrict__ xb,
                      const __bf16* __restrict__ wqt, const __bf16* __restrict__ wkt,
                      const __bf16* __restrict__ wvt,
                      __bf16* __restrict__ qb, __bf16* __restrict__ kbuf,
                      __bf16* __restrict__ vtb) {
  int lane = threadIdx.x & 31;
  int m0 = blockIdx.x * 16;
  int n0 = blockIdx.y * 16;
  int z  = blockIdx.z;
  const __bf16* wt = (z == 0) ? wqt : (z == 1) ? wkt : wvt;
  f32x8 acc = {};
#pragma unroll
  for (int s = 0; s < EMB / 32; ++s) {
    bf16x16 a = load_a_bf16(xb, EMB, m0, s * 32, lane);
    bf16x16 b = load_b_bf16(wt, EMB, n0, s * 32, lane);
    acc = wmma_bf16(a, b, acc);
  }
  int half = lane >> 4;
  int col  = n0 + (lane & 15);
#pragma unroll
  for (int r = 0; r < 8; ++r) {                 // C/D: VGPR r -> M = r + half*8
    int m = m0 + r + half * 8;
    float v = acc[r];
    if (z == 0)      qb[(size_t)m * EMB + col]   = (__bf16)v;
    else if (z == 1) kbuf[(size_t)m * EMB + col] = (__bf16)v;
    else {                                       // V stored transposed [b][c][t]
      int bb = m >> 11, tt = m & (SEQ - 1);
      vtb[((size_t)bb * EMB + col) * SEQ + tt]   = (__bf16)v;
    }
  }
}

// ---- 5. causal attention, one block per (batch, 16-query tile) --------------
#define SP 2056   // 2048 + 8 pad to skew LDS banks
__global__ void k_attn(const __bf16* __restrict__ qb, const __bf16* __restrict__ kbuf,
                       const __bf16* __restrict__ vtb, __bf16* __restrict__ ob) {
  __shared__ float S[16][SP];   // 131.6 KB of the 320 KB/WGP LDS
  __shared__ float sm_l[16];
  int qt = blockIdx.x, bidx = blockIdx.y;
  int q0 = qt * 16;
  int tid = threadIdx.x;
  int wave = tid >> 5, lane = tid & 31, half = lane >> 4;
  const float scale = 0.05103103630798287f;     // 1/sqrt(384)

  // phase 1: S = scale * Q K^T with causal mask (waves split key tiles)
  for (int kt = wave; kt <= qt; kt += 8) {      // wave-uniform trip count
    f32x8 acc = {};
#pragma unroll
    for (int s = 0; s < EMB / 32; ++s) {
      bf16x16 a = load_a_bf16(qb + (size_t)(bidx * SEQ + q0) * EMB, EMB, 0, s * 32, lane);
      bf16x16 b = load_b_bf16(kbuf + (size_t)(bidx * SEQ + kt * 16) * EMB, EMB, 0, s * 32, lane);
      acc = wmma_bf16(a, b, acc);
    }
    int key = kt * 16 + (lane & 15);
#pragma unroll
    for (int r = 0; r < 8; ++r) {
      int rr = r + half * 8;
      S[rr][key] = (key <= q0 + rr) ? acc[r] * scale : -INFINITY;
    }
  }
  __syncthreads();

  // phase 2: row softmax (16 threads per row, shfl width-16 reductions)
  int nkb   = (q0 + 16 + 31) >> 5;              // key blocks of 32 needed later
  int ncols = nkb << 5;
  {
    int row = tid >> 4, sub = tid & 15;
    int qabs = q0 + row;
    float m = -INFINITY;
    for (int c = sub; c < ncols; c += 16) {
      float v = (c <= qabs) ? S[row][c] : -INFINITY;
      m = fmaxf(m, v);
    }
#pragma unroll
    for (int msk = 8; msk >= 1; msk >>= 1) m = fmaxf(m, __shfl_xor(m, msk, 16));
    float l = 0.f;
    for (int c = sub; c < ncols; c += 16) {
      float v = (c <= qabs) ? S[row][c] : -INFINITY;
      float p = __expf(v - m);
      S[row][c] = p;                            // also zeroes the masked tail
      l += p;
    }
#pragma unroll
    for (int msk = 8; msk >= 1; msk >>= 1) l += __shfl_xor(l, msk, 16);
    if (sub == 0) sm_l[row] = l;
  }
  __syncthreads();

  // phase 3: O = P V, each wave owns 3 of the 24 column tiles
  f32x8 acc0 = {}, acc1 = {}, acc2 = {};
  const __bf16* vb = vtb + (size_t)bidx * EMB * SEQ;
  for (int kb = 0; kb < nkb; ++kb) {
    bf16x16 a;                                  // P fragment, f32 LDS -> bf16
    {
      int row = lane & 15;
      const float* p = &S[row][kb * 32 + half * 8];
#pragma unroll
      for (int e = 0; e < 8; ++e) {
        a[e]     = (__bf16)p[e];
        a[e + 8] = (__bf16)p[e + 16];
      }
    }
    bf16x16 b0 = load_b_bf16(vb, SEQ, (wave +  0) * 16, kb * 32, lane);
    bf16x16 b1 = load_b_bf16(vb, SEQ, (wave +  8) * 16, kb * 32, lane);
    bf16x16 b2 = load_b_bf16(vb, SEQ, (wave + 16) * 16, kb * 32, lane);
    acc0 = wmma_bf16(a, b0, acc0);
    acc1 = wmma_bf16(a, b1, acc1);
    acc2 = wmma_bf16(a, b2, acc2);
  }
#pragma unroll
  for (int r = 0; r < 8; ++r) {
    int rr = r + half * 8;
    float inv = 1.0f / sm_l[rr];
    size_t rowoff = (size_t)(bidx * SEQ + q0 + rr) * EMB + (lane & 15);
    ob[rowoff + (wave +  0) * 16] = (__bf16)(acc0[r] * inv);
    ob[rowoff + (wave +  8) * 16] = (__bf16)(acc1[r] * inv);
    ob[rowoff + (wave + 16) * 16] = (__bf16)(acc2[r] * inv);
  }
}

// ---- 6. head GEMM + fused partial logsumexp ---------------------------------
// logits = out @ W_head + b_head  (NT stores to d_out: written once, barely
// re-read, so keep the 1.65 GB stream out of L2 and preserve the resident
// bf16 weights). Each block also emits one (max, sumexp) pair per row.
__global__ void k_head(const __bf16* __restrict__ ob, const __bf16* __restrict__ wht,
                       const float* __restrict__ bh, float* __restrict__ out,
                       float* __restrict__ pm, float* __restrict__ pl) {
  __shared__ float sm_m[8][16];   // [wave][row-in-tile]
  __shared__ float sm_s[8][16];
  int lane = threadIdx.x & 31, wave = threadIdx.x >> 5, half = lane >> 4;
  int nt = blockIdx.x * 8 + wave;
  int m0 = blockIdx.y * 16;
  bool valid = (nt < NTHEAD);                   // wave-uniform

  float wm[8], wl[8];
#pragma unroll
  for (int r = 0; r < 8; ++r) { wm[r] = -INFINITY; wl[r] = 0.f; }

  if (valid) {                                  // EXEC stays all-ones inside
    int n0 = nt * 16;
    f32x8 acc = {};
#pragma unroll
    for (int s = 0; s < EMB / 32; ++s) {
      bf16x16 a = load_a_bf16(ob,  EMB, m0, s * 32, lane);
      bf16x16 b = load_b_bf16(wht, EMB, n0, s * 32, lane);
      acc = wmma_bf16(a, b, acc);
    }
    int col = n0 + (lane & 15);
    bool cok = (col < VOCAB);
    float bias = cok ? bh[col] : 0.f;
#pragma unroll
    for (int r = 0; r < 8; ++r) {
      int row = m0 + r + half * 8;
      float v = acc[r] + bias;
      if (cok)
        __builtin_nontemporal_store(v, &out[(size_t)row * VOCAB + col]);
      // per-row (max, sumexp) across the 16 lanes of this half
      float vm = cok ? v : -INFINITY;
      float mr = vm;
#pragma unroll
      for (int msk = 8; msk >= 1; msk >>= 1) mr = fmaxf(mr, __shfl_xor(mr, msk, 16));
      float lr = (vm == -INFINITY) ? 0.f : __expf(vm - mr);
#pragma unroll
      for (int msk = 8; msk >= 1; msk >>= 1) lr += __shfl_xor(lr, msk, 16);
      wm[r] = mr; wl[r] = lr;                   // all 16 lanes hold the result
    }
  }
  if ((lane & 15) == 0) {                       // lane 0 -> rows 0..7, lane 16 -> rows 8..15
#pragma unroll
    for (int r = 0; r < 8; ++r) {
      sm_m[wave][r + half * 8] = wm[r];
      sm_s[wave][r + half * 8] = wl[r];
    }
  }
  __syncthreads();
  if (threadIdx.x < 16) {                       // merge the 8 waves of this block
    int rr = threadIdx.x;
    float m = -INFINITY, l = 0.f;
#pragma unroll
    for (int w = 0; w < 8; ++w) lse_merge(m, l, sm_m[w][rr], sm_s[w][rr]);
    size_t row = (size_t)(m0 + rr);
    pm[row * NBLKX + blockIdx.x] = m;
    pl[row * NBLKX + blockIdx.x] = l;
  }
}

// ---- 7. per-row merge of partials + target gather ---------------------------
__global__ void k_lsered(const float* __restrict__ pm, const float* __restrict__ pl,
                         const float* __restrict__ logits, const int* __restrict__ targets,
                         float* __restrict__ rowloss) {
  __shared__ float smx[128], ssm[128];
  int row = blockIdx.x, tid = threadIdx.x;
  float m = -INFINITY, l = 0.f;
  for (int i = tid; i < NBLKX; i += 128)
    lse_merge(m, l, pm[(size_t)row * NBLKX + i], pl[(size_t)row * NBLKX + i]);
  smx[tid] = m; ssm[tid] = l;
  __syncthreads();
  for (int s = 64; s > 0; s >>= 1) {
    if (tid < s) {
      float mm = smx[tid], ll = ssm[tid];
      lse_merge(mm, ll, smx[tid + s], ssm[tid + s]);
      smx[tid] = mm; ssm[tid] = ll;
    }
    __syncthreads();
  }
  if (tid == 0) {
    float lse = smx[0] + __logf(ssm[0]);
    rowloss[row] = lse - logits[(size_t)row * VOCAB + targets[row]];
  }
}

// ---- 8. mean loss -----------------------------------------------------------
__global__ void k_loss(const float* __restrict__ rowloss, float* __restrict__ out) {
  __shared__ float ss[256];
  int tid = threadIdx.x;
  float s = 0.f;
  for (int i = tid; i < MTOT; i += 256) s += rowloss[i];
  ss[tid] = s; __syncthreads();
  for (int k = 128; k > 0; k >>= 1) { if (tid < k) ss[tid] += ss[tid + k]; __syncthreads(); }
  if (tid == 0) out[(size_t)MTOT * VOCAB] = ss[0] / (float)MTOT;
}

// ---- launch -----------------------------------------------------------------
extern "C" void kernel_launch(void* const* d_in, const int* in_sizes, int n_in,
                              void* d_out, int out_size, void* d_ws, size_t ws_size,
                              hipStream_t stream) {
  const int*   idx = (const int*)d_in[0];
  const int*   tgt = (const int*)d_in[1];
  const float* tok = (const float*)d_in[2];
  const float* pos = (const float*)d_in[3];
  const float* Wk  = (const float*)d_in[4];
  const float* Wq  = (const float*)d_in[5];
  const float* Wv  = (const float*)d_in[6];
  const float* Wh  = (const float*)d_in[7];
  const float* bh  = (const float*)d_in[8];
  float* out = (float*)d_out;

  char* ws = (char*)d_ws;
  size_t off = 0;
  auto alloc = [&](size_t bytes) -> void* {
    void* p = ws + off;
    off = (off + bytes + 255) & ~(size_t)255;
    return p;
  };
  __bf16* xb   = (__bf16*)alloc((size_t)MTOT * EMB * 2);   //  6.3 MB
  __bf16* wqt  = (__bf16*)alloc((size_t)EMB * EMB * 2);
  __bf16* wkt  = (__bf16*)alloc((size_t)EMB * EMB * 2);
  __bf16* wvt  = (__bf16*)alloc((size_t)EMB * EMB * 2);
  __bf16* qb   = (__bf16*)alloc((size_t)MTOT * EMB * 2);
  __bf16* kbuf = (__bf16*)alloc((size_t)MTOT * EMB * 2);
  __bf16* vtb  = (__bf16*)alloc((size_t)NB * EMB * SEQ * 2);
  __bf16* ob   = (__bf16*)alloc((size_t)MTOT * EMB * 2);
  __bf16* wht  = (__bf16*)alloc((size_t)VPAD * EMB * 2);   // 38.6 MB
  float*  pm   = (float*)alloc((size_t)MTOT * NBLKX * 4);  // 12.9 MB
  float*  pl   = (float*)alloc((size_t)MTOT * NBLKX * 4);  // 12.9 MB
  float*  rowloss = (float*)alloc((size_t)MTOT * 4);
  (void)ws_size; (void)in_sizes; (void)n_in; (void)out_size;

  k_embed <<<(MTOT * EMB + 255) / 256, 256, 0, stream>>>(idx, tok, pos, xb);
  k_tw    <<<(EMB * EMB + 255) / 256, 256, 0, stream>>>(Wq, wqt);
  k_tw    <<<(EMB * EMB + 255) / 256, 256, 0, stream>>>(Wk, wkt);
  k_tw    <<<(EMB * EMB + 255) / 256, 256, 0, stream>>>(Wv, wvt);
  k_twhead<<<(int)(((size_t)VPAD * EMB + 255) / 256), 256, 0, stream>>>(Wh, wht);
  k_qkv   <<<dim3(MTOT / 16, EMB / 16, 3), 32, 0, stream>>>(xb, wqt, wkt, wvt, qb, kbuf, vtb);
  k_attn  <<<dim3(SEQ / 16, NB), 256, 0, stream>>>(qb, kbuf, vtb, ob);
  k_head  <<<dim3(NBLKX, MTOT / 16), 256, 0, stream>>>(ob, wht, bh, out, pm, pl);
  k_lsered<<<MTOT, 128, 0, stream>>>(pm, pl, out, tgt, rowloss);
  k_loss  <<<1, 256, 0, stream>>>(rowloss, out);
}
